// Attention_11845519802454
// MI455X (gfx1250) — matrix operand order repaired and checked by reading, not compiled, not run on hardware
//
#include <hip/hip_runtime.h>
#include <hip/hip_bf16.h>
#include <math.h>

// ---------------------------------------------------------------------------
// Attention block (GroupNorm -> 1x1 QKV -> softmax attention -> 1x1 proj + res)
// b=32, c=256, h=w=32 (n=1024).  All GEMMs on v_wmma_f32_16x16x32_f16.
// Workspace (f16): xnT | Qt | Kt | Vc | Ot | wqkv_h | wout_h  = ~80.5 MB.
// ---------------------------------------------------------------------------

#define BB   32
#define CC   256
#define NN   1024
#define NG   8
#define CPG  32          // channels per group
#define SCALE 0.0625f    // 256^-0.5

typedef __attribute__((ext_vector_type(16))) _Float16 v16h;
typedef __attribute__((ext_vector_type(8)))  _Float16 v8h;
typedef __attribute__((ext_vector_type(8)))  float    v8f;

// D = A(16x32 f16) * B(32x16 f16) + C(16x16 f32)
__device__ __forceinline__ v8f wmma16(v16h a, v16h b, v8f c) {
  return __builtin_amdgcn_wmma_f32_16x16x32_f16(false, a, false, b,
                                                (short)0, c, false, false);
}

// ---- A fragment, 16x32 f16, source = f16 row-major matrix ------------------
// ISA layout: lane m=l&15; lanes<16 hold K 0..7 & 16..23, lanes>=16 K 8..15 & 24..31
__device__ __forceinline__ v16h load_a_h(const _Float16* M, int ldm,
                                         int m0, int k0, int lane) {
  int m    = lane & 15;
  int koff = (lane >> 4) << 3;            // 0 or 8
  const _Float16* p = M + (size_t)(m0 + m) * ldm + k0 + koff;
  v8h lo = *(const v8h*)p;                // 16B aligned
  v8h hi = *(const v8h*)(p + 16);
  v16h a;
#pragma unroll
  for (int j = 0; j < 8; ++j) { a[j] = lo[j]; a[8 + j] = hi[j]; }
  return a;
}

// ---- A fragment from f16 row-major matrix, scaled (used for Q*c^-0.5) ------
__device__ __forceinline__ v16h load_a_h_scaled(const _Float16* M, int ldm,
                                                int m0, int k0, int lane,
                                                float scale) {
  int m    = lane & 15;
  int koff = (lane >> 4) << 3;
  const _Float16* p = M + (size_t)(m0 + m) * ldm + k0 + koff;
  v8h lo = *(const v8h*)p;
  v8h hi = *(const v8h*)(p + 16);
  v16h a;
#pragma unroll
  for (int j = 0; j < 8; ++j) {
    a[j]     = (_Float16)((float)lo[j] * scale);
    a[8 + j] = (_Float16)((float)hi[j] * scale);
  }
  return a;
}

// ---- B fragment, 32x16 f16.  Source M is stored "N-major": M[n][k] --------
// ISA layout: lane n=l&15; lanes<16 hold K 0..7 & 16..23, lanes>=16 K 8..15 &
// 24..31 -> two 8-half (16B) loads per lane.
__device__ __forceinline__ v16h load_b_h(const _Float16* M, int ldm,
                                         int n0, int k0, int lane) {
  int n  = lane & 15;
  int kb = (lane >> 4) << 3;              // 0 or 8
  const _Float16* p = M + (size_t)(n0 + n) * ldm + k0 + kb;
  v8h lo = *(const v8h*)p;
  v8h hi = *(const v8h*)(p + 16);
  v16h b;
#pragma unroll
  for (int j = 0; j < 8; ++j) { b[j] = lo[j]; b[8 + j] = hi[j]; }
  return b;
}

// cross-lane reductions within 16-lane halves (rows of a C/D tile)
__device__ __forceinline__ float rmax16(float v) {
  v = fmaxf(v, __shfl_xor(v, 1, 32));
  v = fmaxf(v, __shfl_xor(v, 2, 32));
  v = fmaxf(v, __shfl_xor(v, 4, 32));
  v = fmaxf(v, __shfl_xor(v, 8, 32));
  return v;
}
__device__ __forceinline__ float rsum16(float v) {
  v += __shfl_xor(v, 1, 32);
  v += __shfl_xor(v, 2, 32);
  v += __shfl_xor(v, 4, 32);
  v += __shfl_xor(v, 8, 32);
  return v;
}

// ---------------------------------------------------------------------------
// Kernel 0: one-shot fp32 -> f16 weight conversion.
// ---------------------------------------------------------------------------
__global__ __launch_bounds__(256) void cvt_kernel(const float* __restrict__ src,
                                                  _Float16* __restrict__ dst,
                                                  int n) {
  int i = blockIdx.x * 256 + threadIdx.x;
  if (i < n) dst[i] = (_Float16)src[i];
}

// ---------------------------------------------------------------------------
// Kernel 1: GroupNorm.  One block per (batch, group).  Output f16 xnT[b][n][c].
// ---------------------------------------------------------------------------
__global__ __launch_bounds__(256) void gn_kernel(const float* __restrict__ x,
                                                 const float* __restrict__ gw,
                                                 const float* __restrict__ gb,
                                                 _Float16* __restrict__ xnT) {
  __shared__ float rs[256], rs2[256];
  int bg  = blockIdx.x;
  int bb  = bg >> 3, g = bg & 7;
  int tid = threadIdx.x;
  const float* xp = x + ((size_t)bb * CC + g * CPG) * NN;

  float s = 0.f, s2 = 0.f;
  for (int i = tid; i < CPG * NN; i += 256) {
    float v = xp[i];
    s += v; s2 += v * v;
  }
  rs[tid] = s; rs2[tid] = s2;
  __syncthreads();
  for (int off = 128; off; off >>= 1) {
    if (tid < off) { rs[tid] += rs[tid + off]; rs2[tid] += rs2[tid + off]; }
    __syncthreads();
  }
  const float inv = 1.f / (CPG * NN);
  float mean = rs[0] * inv;
  float var  = rs2[0] * inv - mean * mean;
  float rstd = rsqrtf(var + 1e-5f);

  for (int ch = 0; ch < CPG; ++ch) {
    int cg = g * CPG + ch;
    float w = gw[cg], b = gb[cg];
    const float* xr = x + ((size_t)bb * CC + cg) * NN;
    _Float16* dst = xnT + (size_t)bb * NN * CC + cg;
    for (int p = tid; p < NN; p += 256)
      dst[(size_t)p * CC] = (_Float16)((xr[p] - mean) * rstd * w + b);
  }
}

// ---------------------------------------------------------------------------
// Kernel 2: QKV GEMM.  qkv[o][pix] = sum_c wqkv[o][c] * xn[c][pix].
// One wave per 16x16 output tile. Writes Qt/Kt [b][n][c], Vc [b][c][n].
// ---------------------------------------------------------------------------
__global__ __launch_bounds__(256) void qkv_kernel(const _Float16* __restrict__ wqkv,
                                                  const _Float16* __restrict__ xnT,
                                                  _Float16* __restrict__ Qt,
                                                  _Float16* __restrict__ Kt,
                                                  _Float16* __restrict__ Vc) {
  int wid  = blockIdx.x * 8 + (threadIdx.x >> 5);
  int lane = threadIdx.x & 31;
  int pt = wid & 63;
  int t2 = wid >> 6;
  int ot = t2 % 48;          // 48 row tiles of 768
  int bb = t2 / 48;
  const _Float16* xb = xnT + (size_t)bb * NN * CC;

  v8f acc = 0.0f;
#pragma unroll
  for (int t = 0; t < 8; ++t) {
    v16h a = load_a_h(wqkv, CC, ot * 16, t * 32, lane);
    v16h b = load_b_h(xb, CC, pt * 16, t * 32, lane);
    acc = wmma16(a, b, acc);
  }

  int base = (lane >> 4) << 3;            // 0 or 8 (D rows)
  int pix  = pt * 16 + (lane & 15);
  if (ot < 16) {                          // Q -> Qt[b][pix][c]
    v8h h;
#pragma unroll
    for (int v = 0; v < 8; ++v) h[v] = (_Float16)acc[v];
    *(v8h*)(Qt + ((size_t)bb * NN + pix) * CC + ot * 16 + base) = h;
  } else if (ot < 32) {                   // K -> Kt[b][pix][c]
    v8h h;
#pragma unroll
    for (int v = 0; v < 8; ++v) h[v] = (_Float16)acc[v];
    *(v8h*)(Kt + ((size_t)bb * NN + pix) * CC + (ot - 16) * 16 + base) = h;
  } else {                                // V -> Vc[b][c][pix]
    int o0 = (ot - 32) * 16 + base;
#pragma unroll
    for (int v = 0; v < 8; ++v)
      Vc[((size_t)bb * CC + o0 + v) * NN + pix] = (_Float16)acc[v];
  }
}

// ---------------------------------------------------------------------------
// Kernel 3: flash attention.  One wave owns a 16-query tile; streams keys in
// chunks of 32; O accumulator = 16 f32 C/D tiles (16q x 256c).
// ---------------------------------------------------------------------------
__global__ __launch_bounds__(128) void attn_kernel(const _Float16* __restrict__ Qt,
                                                   const _Float16* __restrict__ Kt,
                                                   const _Float16* __restrict__ Vc,
                                                   _Float16* __restrict__ Ot) {
  __shared__ _Float16 lds_p[4][16 * 32];  // per-wave P staging (1 KB each)
  int wave = threadIdx.x >> 5;
  int lane = threadIdx.x & 31;
  int gw   = blockIdx.x * 4 + wave;
  int qt   = gw & 63;
  int bb   = gw >> 6;

  const _Float16* Qb = Qt + (size_t)bb * NN * CC;
  const _Float16* Kb = Kt + (size_t)bb * NN * CC;
  const _Float16* Vb = Vc + (size_t)bb * CC * NN;
  int q0 = qt * 16;

  // Q fragments over full c, scale folded in
  v16h qa[8];
#pragma unroll
  for (int t = 0; t < 8; ++t)
    qa[t] = load_a_h_scaled(Qb, CC, q0, t * 32, lane, SCALE);

  float rmax[8], rsum[8];
  v8f oacc[16];
#pragma unroll
  for (int v = 0; v < 8; ++v) { rmax[v] = -3.0e38f; rsum[v] = 0.f; }
#pragma unroll
  for (int ct = 0; ct < 16; ++ct) oacc[ct] = 0.0f;

  int base = (lane >> 4) << 3;            // C/D row base
  int coln = lane & 15;
  _Float16* lp = lds_p[wave];

  for (int kc = 0; kc < NN; kc += 32) {
    // S tiles: 16q x 32k
    v8f s0 = 0.0f, s1 = 0.0f;
#pragma unroll
    for (int t = 0; t < 8; ++t) {
      v16h b0 = load_b_h(Kb, CC, kc,      t * 32, lane);
      v16h b1 = load_b_h(Kb, CC, kc + 16, t * 32, lane);
      s0 = wmma16(qa[t], b0, s0);
      s1 = wmma16(qa[t], b1, s1);
    }

    // online softmax update (per-row stats live replicated in 16-lane halves)
    float fscale[8];
#pragma unroll
    for (int v = 0; v < 8; ++v) {
      float m = rmax16(fmaxf(s0[v], s1[v]));
      float mnew = fmaxf(rmax[v], m);
      float f  = __expf(rmax[v] - mnew);
      float p0 = __expf(s0[v] - mnew);
      float p1 = __expf(s1[v] - mnew);
      rsum[v] = rsum[v] * f + rsum16(p0 + p1);
      rmax[v] = mnew;
      fscale[v] = f;
      s0[v] = p0; s1[v] = p1;
    }
#pragma unroll
    for (int ct = 0; ct < 16; ++ct)
#pragma unroll
      for (int v = 0; v < 8; ++v) oacc[ct][v] *= fscale[v];

    // P: C/D layout -> A layout via per-wave LDS tile
#pragma unroll
    for (int v = 0; v < 8; ++v) {
      lp[(base + v) * 32 + coln]      = (_Float16)s0[v];
      lp[(base + v) * 32 + 16 + coln] = (_Float16)s1[v];
    }
    asm volatile("s_wait_dscnt 0x0" ::: "memory");
    int m    = lane & 15;
    int koff = (lane >> 4) << 3;
    v8h lo = *(const v8h*)(lp + m * 32 + koff);
    v8h hi = *(const v8h*)(lp + m * 32 + koff + 16);
    v16h pa;
#pragma unroll
    for (int j = 0; j < 8; ++j) { pa[j] = lo[j]; pa[8 + j] = hi[j]; }

    // O += P * V  over all 256 output channels
#pragma unroll
    for (int ct = 0; ct < 16; ++ct) {
      v16h bv = load_b_h(Vb, NN, ct * 16, kc, lane);
      oacc[ct] = wmma16(pa, bv, oacc[ct]);
    }
  }

  // normalize and store Ot[b][q][c]
  _Float16* ob = Ot + (size_t)bb * NN * CC;
#pragma unroll
  for (int v = 0; v < 8; ++v) rsum[v] = 1.f / rsum[v];
#pragma unroll
  for (int ct = 0; ct < 16; ++ct)
#pragma unroll
    for (int v = 0; v < 8; ++v)
      ob[(size_t)(q0 + base + v) * CC + ct * 16 + coln] =
          (_Float16)(oacc[ct][v] * rsum[v]);
}

// ---------------------------------------------------------------------------
// Kernel 4: out-proj + bias + residual.  y[o][pix] = wout[o][:]·Ot[pix][:]
// ---------------------------------------------------------------------------
__global__ __launch_bounds__(256) void proj_kernel(const _Float16* __restrict__ wout,
                                                   const float* __restrict__ bout,
                                                   const _Float16* __restrict__ Ot,
                                                   const float* __restrict__ x,
                                                   float* __restrict__ out) {
  int wid  = blockIdx.x * 8 + (threadIdx.x >> 5);
  int lane = threadIdx.x & 31;
  int pt = wid & 63;
  int t2 = wid >> 6;
  int ot = t2 & 15;
  int bb = t2 >> 4;
  const _Float16* ob = Ot + (size_t)bb * NN * CC;

  v8f acc = 0.0f;
#pragma unroll
  for (int t = 0; t < 8; ++t) {
    v16h a = load_a_h(wout, CC, ot * 16, t * 32, lane);
    v16h b = load_b_h(ob, CC, pt * 16, t * 32, lane);
    acc = wmma16(a, b, acc);
  }

  int base = (lane >> 4) << 3;
  int pix  = pt * 16 + (lane & 15);
#pragma unroll
  for (int v = 0; v < 8; ++v) {
    int o = ot * 16 + base + v;
    size_t idx = ((size_t)bb * CC + o) * NN + pix;
    out[idx] = acc[v] + bout[o] + x[idx];
  }
}

// ---------------------------------------------------------------------------
extern "C" void kernel_launch(void* const* d_in, const int* in_sizes, int n_in,
                              void* d_out, int out_size, void* d_ws, size_t ws_size,
                              hipStream_t stream) {
  const float* x    = (const float*)d_in[0];
  const float* gnw  = (const float*)d_in[1];
  const float* gnb  = (const float*)d_in[2];
  const float* wqkv = (const float*)d_in[3];
  const float* wout = (const float*)d_in[4];
  const float* bout = (const float*)d_in[5];
  float* out = (float*)d_out;

  const size_t plane = (size_t)BB * NN * CC;   // 8M halfs = 16 MB
  _Float16* xnT    = (_Float16*)d_ws;
  _Float16* Qt     = xnT + plane;
  _Float16* Kt     = Qt + plane;
  _Float16* Vc     = Kt + plane;
  _Float16* Ot     = Vc + plane;
  _Float16* wqkv_h = Ot + plane;               // 768*256 halfs
  _Float16* wout_h = wqkv_h + 768 * CC;        // 256*256 halfs

  cvt_kernel <<<(768 * CC) / 256,   256, 0, stream>>>(wqkv, wqkv_h, 768 * CC);
  cvt_kernel <<<(CC * CC) / 256,    256, 0, stream>>>(wout, wout_h, CC * CC);
  gn_kernel  <<<BB * NG,            256, 0, stream>>>(x, gnw, gnb, xnT);
  qkv_kernel <<<BB * 48 * 64 / 8,   256, 0, stream>>>(wqkv_h, xnT, Qt, Kt, Vc);
  attn_kernel<<<BB * 64 / 4,        128, 0, stream>>>(Qt, Kt, Vc, Ot);
  proj_kernel<<<BB * 16 * 64 / 8,   256, 0, stream>>>(wout_h, bout, Ot, x, out);
}
